// MultiHeadSelfAttention_12713103196901
// MI455X (gfx1250) — matrix operand order repaired and checked by reading, not compiled, not run on hardware
//
#include <hip/hip_runtime.h>
#include <hip/hip_bf16.h>

typedef __attribute__((ext_vector_type(16))) _Float16 v16h;
typedef __attribute__((ext_vector_type(8)))  _Float16 v8h;
typedef __attribute__((ext_vector_type(4)))  _Float16 v4h;
typedef __attribute__((ext_vector_type(8)))  float    v8f;
typedef __attribute__((ext_vector_type(4)))  float    v4f;

#define H_HEADS 16
#define D_HEAD  64
#define S_LEN   2048
#define C_DIM   1024
#define ROWS_TOT 4096   // B*S

__device__ __forceinline__ v16h cat16(v8h lo, v8h hi) {
    v16h r;
#pragma unroll
    for (int e = 0; e < 8; ++e) { r[e] = lo[e]; r[8 + e] = hi[e]; }
    return r;
}

// A-fragment (16x32 f16) from row-major f16, row stride = rowp's row.
// lane -> row (lane&15); elements 0..7 = K half*8+e, 8..15 = K 16+half*8+e.
__device__ __forceinline__ v16h loadA_f16(const _Float16* __restrict__ rowp, int half) {
    v8h a0 = *(const v8h*)(rowp + half * 8);
    v8h a1 = *(const v8h*)(rowp + 16 + half * 8);
    return cat16(a0, a1);
}

// CDNA5 async copy: 16B global -> LDS, tracked with ASYNCcnt.
// lds_off = wave-relative LDS byte address (= low 32 bits of generic pointer,
// per ISA flat-aperture mapping).
__device__ __forceinline__ void async_b128(const void* lds, const void* g) {
    unsigned      l = (unsigned)(unsigned long long)(size_t)lds;
    unsigned long long a = (unsigned long long)(size_t)g;
    asm volatile("global_load_async_to_lds_b128 %0, %1, off"
                 :: "v"(l), "v"(a) : "memory");
}
__device__ __forceinline__ void wait_async0() {
    asm volatile("s_wait_asynccnt 0x0" ::: "memory");
}
__device__ __forceinline__ void wait_ds0() {
    asm volatile("s_wait_dscnt 0x0" ::: "memory");
}

// ---------------------------------------------------------------------------
// Pre-pass 1: X f32 -> f16 (row-major, unchanged layout)
// ---------------------------------------------------------------------------
__global__ __launch_bounds__(256) void convert_x(
    const float* __restrict__ x, _Float16* __restrict__ xh)
{
    int i = (blockIdx.x * 256 + threadIdx.x) * 4;
    v4f v = *(const v4f*)(x + i);
    v4h o;
#pragma unroll
    for (int e = 0; e < 4; ++e) o[e] = (_Float16)v[e];
    *(v4h*)(xh + i) = o;
}

// ---------------------------------------------------------------------------
// Pre-pass 2: W (K=1024 x N, f32 row-major) -> WT (N x 1024, f16) so GEMM
// B-tiles are K-contiguous straight copies (async-to-LDS friendly).
// ---------------------------------------------------------------------------
__global__ __launch_bounds__(256) void transpose_w(
    const float* __restrict__ W, _Float16* __restrict__ WT, int N)
{
    __shared__ float tile[32][33];
    const int n0 = blockIdx.x * 32;
    const int k0 = blockIdx.y * 32;
    const int tx = threadIdx.x & 31;
    const int ty = threadIdx.x >> 5;      // 0..7
#pragma unroll
    for (int i = 0; i < 4; ++i)
        tile[ty + i * 8][tx] = W[(size_t)(k0 + ty + i * 8) * N + n0 + tx];
    __syncthreads();
#pragma unroll
    for (int i = 0; i < 4; ++i)
        WT[(size_t)(n0 + ty + i * 8) * C_DIM + k0 + tx] =
            (_Float16)tile[tx][ty + i * 8];
}

// ---------------------------------------------------------------------------
// Projection GEMM: out = Xh(4096x1024,f16) @ WT'(1024xN) + bias.
// Block tile 128x128, 8 waves (4M x 2N), wave = 32x64 = 8 WMMA accums.
// W tiles staged by async-to-LDS, double buffered, 1 barrier / K-step.
// vmode=0: N=2048, col<1024 -> qh [s][d], else -> kh [s][d].
// vmode=1: N=1024 -> vhT transposed [bh][d][s] (so attention V staging is a
//          straight copy).
// ---------------------------------------------------------------------------
__global__ __launch_bounds__(256) void proj_gemm(
    const _Float16* __restrict__ Xh, const _Float16* __restrict__ WT,
    const float* __restrict__ bias,
    _Float16* __restrict__ out0, _Float16* __restrict__ out1, int vmode)
{
    __shared__ _Float16 wt[2][128][40];    // [buf][n][k], 80B rows (16B aligned)

    const int tid  = threadIdx.x;
    const int lane = tid & 31;
    const int wave = tid >> 5;
    const int wm   = wave >> 1;            // 0..3
    const int wn   = wave & 1;             // 0..1
    const int half = lane >> 4;
    const int l16  = lane & 15;

    const int row0  = blockIdx.x * 128;
    const int col0  = blockIdx.y * 128;
    const int arow0 = row0 + wm * 32;

    v8f acc[2][4];
#pragma unroll
    for (int i = 0; i < 2; ++i)
#pragma unroll
        for (int j = 0; j < 4; ++j)
#pragma unroll
            for (int e = 0; e < 8; ++e) acc[i][j][e] = 0.0f;

    // stage one 128x32 f16 W tile: 512 x b128, 2 per thread
    auto stage = [&](int buf, int k0) {
#pragma unroll
        for (int it = 0; it < 2; ++it) {
            int slot = it * 256 + tid;     // 0..511
            int n    = slot >> 2;          // 0..127
            int p    = (slot & 3) * 8;     // halves
            async_b128(&wt[buf][n][p], WT + (size_t)(col0 + n) * C_DIM + k0 + p);
        }
    };

    stage(0, 0);
    wait_async0();
    __syncthreads();

    for (int kb = 0; kb < C_DIM / 32; ++kb) {
        const int cur = kb & 1;
        if (kb + 1 < C_DIM / 32) stage(cur ^ 1, (kb + 1) * 32);
        const int k0 = kb * 32;

        v16h a[2];
#pragma unroll
        for (int i = 0; i < 2; ++i)
            a[i] = loadA_f16(Xh + (size_t)(arow0 + i * 16 + l16) * C_DIM + k0, half);

#pragma unroll
        for (int j = 0; j < 4; ++j) {
            int nb = wn * 64 + j * 16 + l16;
            v8h b0 = *(const v8h*)&wt[cur][nb][half * 8];
            v8h b1 = *(const v8h*)&wt[cur][nb][16 + half * 8];
            v16h bb = cat16(b0, b1);
#pragma unroll
            for (int i = 0; i < 2; ++i)
                acc[i][j] = __builtin_amdgcn_wmma_f32_16x16x32_f16(
                    false, a[i], false, bb, (short)0, acc[i][j], false, false);
        }

        wait_async0();
        __syncthreads();
    }

    // epilogue: bias + f16 scatter into head-major layouts
#pragma unroll
    for (int i = 0; i < 2; ++i) {
#pragma unroll
        for (int j = 0; j < 4; ++j) {
            int gcol = col0 + wn * 64 + j * 16 + l16;
            float bv = bias[gcol];
#pragma unroll
            for (int r = 0; r < 8; ++r) {
                int grow = arow0 + i * 16 + half * 8 + r;
                float val = acc[i][j][r] + bv;
                _Float16 hv = (_Float16)val;
                int b = grow >> 11;
                int s = grow & 2047;
                if (vmode == 0) {
                    if (gcol < C_DIM) {
                        int h = gcol >> 6, d = gcol & 63;
                        out0[(((size_t)b * H_HEADS + h) * S_LEN + s) * D_HEAD + d] = hv;
                    } else {
                        int cc = gcol - C_DIM;
                        int h = cc >> 6, d = cc & 63;
                        out1[(((size_t)b * H_HEADS + h) * S_LEN + s) * D_HEAD + d] = hv;
                    }
                } else {
                    int h = gcol >> 6, d = gcol & 63;
                    // transposed V: [bh][d][s]
                    out0[(((size_t)b * H_HEADS + h) * D_HEAD + d) * S_LEN + s] = hv;
                }
            }
        }
    }
}

// ---------------------------------------------------------------------------
// Flash attention: block = (b,h) x 64 q-rows, 4 waves x 16 q-rows.
// 32-key blocks; K & V tiles staged via async-to-LDS, double buffered,
// 1 barrier / key block.  Scores: 4 WMMAs, output: 4 WMMAs per block.
// ---------------------------------------------------------------------------
__global__ __launch_bounds__(128) void attn_kernel(
    const _Float16* __restrict__ qh, const _Float16* __restrict__ kh,
    const _Float16* __restrict__ vhT, float* __restrict__ out)
{
    __shared__ _Float16 kt[2][32][72];     // [buf][key][d], 144B rows
    __shared__ _Float16 vt[2][64][40];     // [buf][d][key], 80B rows
    __shared__ _Float16 pt[4][16][40];     // per-wave P tile [q][key]

    const int bh    = blockIdx.x;          // 0..31
    const int b     = bh >> 4;
    const int h     = bh & 15;
    const int qbase = blockIdx.y * 64;
    const int tid   = threadIdx.x;
    const int lane  = tid & 31;
    const int wave  = tid >> 5;
    const int half  = lane >> 4;
    const int l16   = lane & 15;

    const _Float16* Qp  = qh  + (size_t)bh * S_LEN * D_HEAD;
    const _Float16* Kp  = kh  + (size_t)bh * S_LEN * D_HEAD;
    const _Float16* VpT = vhT + (size_t)bh * D_HEAD * S_LEN;

    // Q fragments, held in registers
    const int qrowA = qbase + wave * 16 + l16;
    v16h qa[2];
#pragma unroll
    for (int c = 0; c < 2; ++c) {
        const _Float16* qp = Qp + (size_t)qrowA * D_HEAD + c * 32;
        qa[c] = cat16(*(const v8h*)(qp + half * 8),
                      *(const v8h*)(qp + 16 + half * 8));
    }

    v8f o[4];
#pragma unroll
    for (int j = 0; j < 4; ++j)
#pragma unroll
        for (int e = 0; e < 8; ++e) o[j][e] = 0.0f;

    float mrow[8], lrow[8];
#pragma unroll
    for (int r = 0; r < 8; ++r) { mrow[r] = -1e30f; lrow[r] = 0.0f; }

    // stage K(32x64) + V(64x32) tiles for one key block: 512 x b128, 4/thread
    auto stage = [&](int buf, int key0) {
#pragma unroll
        for (int it = 0; it < 2; ++it) {
            int slot = it * 128 + tid;     // 0..255
            int key  = slot >> 3;          // 0..31
            int pk   = (slot & 7) * 8;     // 0..56
            async_b128(&kt[buf][key][pk], Kp + (size_t)(key0 + key) * D_HEAD + pk);
            int d    = slot >> 2;          // 0..63
            int pv   = (slot & 3) * 8;     // 0..24
            async_b128(&vt[buf][d][pv], VpT + (size_t)d * S_LEN + key0 + pv);
        }
    };

    const int nkb = qbase / 32 + 2;        // covers keys [0, qbase+63]

    stage(0, 0);
    wait_async0();
    __syncthreads();

    for (int kb = 0; kb < nkb; ++kb) {
        const int cur  = kb & 1;
        const int key0 = kb * 32;
        if (kb + 1 < nkb) stage(cur ^ 1, (kb + 1) * 32);

        // scores: two 16-key tiles, accumulated over two 32-d chunks
        v8f sacc[2];
#pragma unroll
        for (int t = 0; t < 2; ++t) {
            const _Float16* kp = &kt[cur][t * 16 + l16][0];
            v8f a;
#pragma unroll
            for (int e = 0; e < 8; ++e) a[e] = 0.0f;
#pragma unroll
            for (int c = 0; c < 2; ++c) {
                v16h bb = cat16(*(const v8h*)(kp + c * 32 + half * 8),
                                *(const v8h*)(kp + c * 32 + 16 + half * 8));
                a = __builtin_amdgcn_wmma_f32_16x16x32_f16(
                        false, qa[c], false, bb, (short)0, a, false, false);
            }
            sacc[t] = a;
        }

        // online softmax: 16-lane half-wave row reductions
        float pr[2][8];
#pragma unroll
        for (int r = 0; r < 8; ++r) {
            int qg = qbase + wave * 16 + half * 8 + r;
            float mx = -1e30f;
#pragma unroll
            for (int t = 0; t < 2; ++t) {
                int kg = key0 + t * 16 + l16;
                float s = sacc[t][r] * 0.125f;     // 1/sqrt(64)
                if (kg > qg) s = -1e30f;           // causal mask
                pr[t][r] = s;
                mx = fmaxf(mx, s);
            }
#pragma unroll
            for (int m = 1; m < 16; m <<= 1)
                mx = fmaxf(mx, __shfl_xor(mx, m, 32));
            float mnew = fmaxf(mrow[r], mx);
            float corr = __expf(mrow[r] - mnew);
            float psum = 0.0f;
#pragma unroll
            for (int t = 0; t < 2; ++t) {
                float p = __expf(pr[t][r] - mnew);
                pr[t][r] = p;
                psum += p;
            }
#pragma unroll
            for (int m = 1; m < 16; m <<= 1)
                psum += __shfl_xor(psum, m, 32);
            lrow[r] = lrow[r] * corr + psum;
            mrow[r] = mnew;
#pragma unroll
            for (int j = 0; j < 4; ++j) o[j][r] *= corr;
        }

        // re-layout P (C/D -> A fragment) through per-wave LDS tile;
        // intra-wave LDS is in-order, s_wait_dscnt is enough (no barrier)
#pragma unroll
        for (int t = 0; t < 2; ++t)
#pragma unroll
            for (int r = 0; r < 8; ++r)
                pt[wave][half * 8 + r][t * 16 + l16] = (_Float16)pr[t][r];
        wait_ds0();
        v16h pa = cat16(*(const v8h*)&pt[wave][l16][half * 8],
                        *(const v8h*)&pt[wave][l16][16 + half * 8]);

        // O += P @ V  (4 d-tiles)
#pragma unroll
        for (int j = 0; j < 4; ++j) {
            v16h bb = cat16(*(const v8h*)&vt[cur][j * 16 + l16][half * 8],
                            *(const v8h*)&vt[cur][j * 16 + l16][16 + half * 8]);
            o[j] = __builtin_amdgcn_wmma_f32_16x16x32_f16(
                       false, pa, false, bb, (short)0, o[j], false, false);
        }

        wait_async0();
        __syncthreads();
    }

    // finalize: divide by row sums, write fp32 [B][S][H*64]
#pragma unroll
    for (int j = 0; j < 4; ++j) {
#pragma unroll
        for (int r = 0; r < 8; ++r) {
            int qg = qbase + wave * 16 + half * 8 + r;
            int d  = j * 16 + l16;
            out[((size_t)b * S_LEN + qg) * C_DIM + h * D_HEAD + d] = o[j][r] / lrow[r];
        }
    }
}

extern "C" void kernel_launch(void* const* d_in, const int* in_sizes, int n_in,
                              void* d_out, int out_size, void* d_ws, size_t ws_size,
                              hipStream_t stream) {
    const float* x   = (const float*)d_in[0];   // (2,2048,1024)
    const float* Wqk = (const float*)d_in[1];   // (1024,2048)
    const float* bqk = (const float*)d_in[2];   // (2048)
    const float* Wv  = (const float*)d_in[3];   // (1024,1024)
    const float* bv  = (const float*)d_in[4];   // (1024)
    float* out = (float*)d_out;                 // (2,2048,1024) f32

    // f16 workspace (halves): xh | wqkT | wvT | qh | kh | vhT  (~38 MB)
    _Float16* xh   = (_Float16*)d_ws;
    _Float16* wqkT = xh   + (size_t)ROWS_TOT * C_DIM;        // 4 Mi
    _Float16* wvT  = wqkT + (size_t)2048 * C_DIM;            // 2 Mi
    _Float16* qh   = wvT  + (size_t)1024 * C_DIM;            // 1 Mi
    _Float16* kh   = qh   + (size_t)ROWS_TOT * C_DIM;
    _Float16* vhT  = kh   + (size_t)ROWS_TOT * C_DIM;

    convert_x<<<dim3(ROWS_TOT * C_DIM / 1024), 256, 0, stream>>>(x, xh);
    transpose_w<<<dim3(2048 / 32, C_DIM / 32), 256, 0, stream>>>(Wqk, wqkT, 2048);
    transpose_w<<<dim3(1024 / 32, C_DIM / 32), 256, 0, stream>>>(Wv,  wvT,  1024);

    proj_gemm<<<dim3(ROWS_TOT / 128, 2048 / 128), 256, 0, stream>>>(
        xh, wqkT, bqk, qh, kh, 0);
    proj_gemm<<<dim3(ROWS_TOT / 128, 1024 / 128), 256, 0, stream>>>(
        xh, wvT, bv, vhT, nullptr, 1);

    attn_kernel<<<dim3(2 * H_HEADS, S_LEN / 64), 128, 0, stream>>>(
        qh, kh, vhT, out);
}
